// GNN_4217657884933
// MI455X (gfx1250) — matrix operand order repaired and checked by reading, not compiled, not run on hardware
//
#include <hip/hip_runtime.h>
#include <cstdint>
#include <cstddef>

// ============================================================================
// MI455X (gfx1250) GNN forward.
//  - All dense linears run through v_wmma_f32_16x16x32_f16 (f16 in, f32 acc).
//  - Each wave register-blocks a 32x64 output tile (2x4 WMMA tiles): A frags
//    reused 4x, B frags 2x -> 24 fresh B/lane per WMMA instead of 64.
//  - Activation f16 mirrors are produced inside GEMM / LN epilogues.
//  - Edge message GEMM fuses gelu * per-head attention * atomic scatter-add.
//  - Segment sums (graph pooling, edge aggregation) use global f32 atomics.
// ============================================================================

typedef _Float16 f16;
typedef _Float16 v16h __attribute__((ext_vector_type(16)));
typedef _Float16 v8h  __attribute__((ext_vector_type(8)));
typedef float    v8f  __attribute__((ext_vector_type(8)));

namespace {

constexpr int N_NODES  = 64000;
constexpr int N_EDGES  = 200000;
constexpr int N_GRAPHS = 1024;
constexpr int W        = 256;
constexpr int W2       = 512;
constexpr int DEPTH    = 2;
constexpr int OUT_X1   = 174;   // sum(NODE_SIZE)

__device__ __forceinline__ float gelu_f(float x) {
  return 0.5f * x * (1.0f + erff(x * 0.70710678118654752440f));
}

// ---------------------------------------------------------------------------
// WMMA GEMM, 32x64 tile per wave (2 M-tiles x 4 N-tiles):
//   out[M,ldc] = epilogue( A[M,K](f16) @ Wt[NoutPad,K]^T (f16) + bias )
// Wt is stored transposed ([Nout][K]) so each B column is a contiguous K run,
// matching the ISA B-matrix lane layout (lanes 0-15: K=k..k+7,k+16..k+23 of col n,
// lanes 16-31: K=k+8..k+15,k+24..k+31).  A follows the 16-bit A 16x32 layout.
// Requires: M % 32 == 0, Ntile % 64 == 0, K % 32 == 0.
// ---------------------------------------------------------------------------
enum { MODE_PLAIN = 0, MODE_GELU = 1, MODE_ADD = 2, MODE_EDGE = 3 };

template <int MODE>
__global__ void __launch_bounds__(256) k_gemm(
    const f16* __restrict__ A, const f16* __restrict__ Wt,
    const float* __restrict__ bias, float* __restrict__ out,
    f16* __restrict__ outh, float* __restrict__ out2,
    const float* __restrict__ att, const int* __restrict__ dsti,
    float* __restrict__ scatter,
    int M, int Ncols, int Ntile, int K, int ldc)
{
  int wave  = (int)((blockIdx.x * blockDim.x + threadIdx.x) >> 5);
  int ntn   = Ntile >> 6;                 // 64-wide N macro tiles
  int total = (M >> 5) * ntn;             // 32-high M macro tiles
  if (wave >= total) return;
  int tm   = (wave / ntn) << 5;
  int tn   = (wave % ntn) << 6;
  int lane = threadIdx.x & 31;
  int hl   = lane >> 4;       // half-wave select
  int l    = lane & 15;

  const f16* arow0 = A  + (size_t)(tm + l) * K;        // A rows tm..tm+15
  const f16* arow1 = arow0 + (size_t)16 * K;           // A rows tm+16..tm+31
  const f16* bcol  = Wt + (size_t)(tn + l) * K;        // B cols tn..tn+63

  v8f acc[2][4] = {};
  for (int k = 0; k < K; k += 32) {
    union { v16h v; v8h h[2]; } a0, a1, b0, b1, b2, b3;
    // A 16x32 f16 layout: lane<16 holds K in {k..k+7} u {k+16..k+23},
    //                     lane>=16 holds K in {k+8..k+15} u {k+24..k+31}
    a0.h[0] = *(const v8h*)(arow0 + k + hl * 8);
    a0.h[1] = *(const v8h*)(arow0 + k + 16 + hl * 8);
    a1.h[0] = *(const v8h*)(arow1 + k + hl * 8);
    a1.h[1] = *(const v8h*)(arow1 + k + 16 + hl * 8);
    // B 32x16 f16 layout (same K order as A): lane<16 -> K in {k..k+7} u {k+16..k+23},
    //                     lane>=16 -> K in {k+8..k+15} u {k+24..k+31}
    b0.h[0] = *(const v8h*)(bcol + k + hl * 8);
    b0.h[1] = *(const v8h*)(bcol + k + 16 + hl * 8);
    b1.h[0] = *(const v8h*)(bcol + (size_t)16 * K + k + hl * 8);
    b1.h[1] = *(const v8h*)(bcol + (size_t)16 * K + k + 16 + hl * 8);
    b2.h[0] = *(const v8h*)(bcol + (size_t)32 * K + k + hl * 8);
    b2.h[1] = *(const v8h*)(bcol + (size_t)32 * K + k + 16 + hl * 8);
    b3.h[0] = *(const v8h*)(bcol + (size_t)48 * K + k + hl * 8);
    b3.h[1] = *(const v8h*)(bcol + (size_t)48 * K + k + 16 + hl * 8);
    acc[0][0] = __builtin_amdgcn_wmma_f32_16x16x32_f16(false, a0.v, false, b0.v, (short)0, acc[0][0], false, false);
    acc[0][1] = __builtin_amdgcn_wmma_f32_16x16x32_f16(false, a0.v, false, b1.v, (short)0, acc[0][1], false, false);
    acc[0][2] = __builtin_amdgcn_wmma_f32_16x16x32_f16(false, a0.v, false, b2.v, (short)0, acc[0][2], false, false);
    acc[0][3] = __builtin_amdgcn_wmma_f32_16x16x32_f16(false, a0.v, false, b3.v, (short)0, acc[0][3], false, false);
    acc[1][0] = __builtin_amdgcn_wmma_f32_16x16x32_f16(false, a1.v, false, b0.v, (short)0, acc[1][0], false, false);
    acc[1][1] = __builtin_amdgcn_wmma_f32_16x16x32_f16(false, a1.v, false, b1.v, (short)0, acc[1][1], false, false);
    acc[1][2] = __builtin_amdgcn_wmma_f32_16x16x32_f16(false, a1.v, false, b2.v, (short)0, acc[1][2], false, false);
    acc[1][3] = __builtin_amdgcn_wmma_f32_16x16x32_f16(false, a1.v, false, b3.v, (short)0, acc[1][3], false, false);
  }

#pragma unroll
  for (int mi = 0; mi < 2; ++mi) {
#pragma unroll
    for (int ni = 0; ni < 4; ++ni) {
      int col    = tn + ni * 16 + l;
      bool colok = col < Ncols;
      float bv   = colok ? bias[col] : 0.0f;
#pragma unroll
      for (int r = 0; r < 8; ++r) {
        int row = tm + mi * 16 + r + (hl << 3);  // C/D: VGPR r -> M=r (+8 hi)
        float v = acc[mi][ni][r] + bv;
        if (MODE == MODE_EDGE) {
          // gelu * per-head attention, atomic scatter-add by destination node
          v = gelu_f(v) * att[row * 4 + (col >> 7)];
          atomicAdd(scatter + (size_t)dsti[row] * ldc + col, v);
        } else if (colok) {
          size_t o = (size_t)row * ldc + col;
          if (MODE == MODE_GELU) v = gelu_f(v);
          if (MODE == MODE_ADD) {
            float nv = out[o] + v;          // xx += xres
            out[o] = nv;
            if (outh) outh[o] = (f16)nv;    // refresh f16 mirror in-epilogue
            if (out2) out2[(size_t)row * Ncols + col] = v;  // residual (x2)
          } else {
            out[o] = v;
            if (outh) outh[o] = (f16)v;
          }
        }
      }
    }
  }
}

// ---- weight transpose + f32->f16: w[K,Nout] -> wt[NoutPad,K] ---------------
__global__ void k_wt(const float* __restrict__ w, f16* __restrict__ wt,
                     int K, int Nout, int NoutPad) {
  int i = blockIdx.x * blockDim.x + threadIdx.x;
  if (i >= K * NoutPad) return;
  int n = i / K, k = i - n * K;
  wt[(size_t)n * K + k] = (n < Nout) ? (f16)w[(size_t)k * Nout + n] : (f16)0.0f;
}

// ---- node embedding: xx = e0[x0] + 0.5 * sum_k e1_k[x_{k+1}+1]*exp(z_k) ----
__global__ void __launch_bounds__(256) k_embed_nodes(
    const int* __restrict__ x, const float* __restrict__ e0,
    const float* t0, const float* t1, const float* t2, const float* t3,
    const float* t4, const float* t5, const float* t6, const float* t7,
    const float* __restrict__ zero,
    float* __restrict__ xx, f16* __restrict__ xxh)
{
  int n = blockIdx.x, c = threadIdx.x;
  const int* xi = x + (size_t)n * 9;
  const float* tabs[8] = {t0, t1, t2, t3, t4, t5, t6, t7};
  float v = e0[(size_t)xi[0] * W + c];
  float s = 0.0f;
#pragma unroll
  for (int k = 0; k < 8; ++k)
    s += tabs[k][(size_t)(xi[1 + k] + 1) * W + c] * expf(zero[k]);
  v += 0.5f * s;
  xx[(size_t)n * W + c]  = v;
  xxh[(size_t)n * W + c] = (f16)v;
}

// ---- rowwise LayerNorm (eps 1e-5, no affine), optional f16 mirror ----------
__global__ void __launch_bounds__(256) k_ln(
    const float* __restrict__ in, float* __restrict__ out, f16* __restrict__ outh)
{
  __shared__ float s1[256], s2[256];
  int n = blockIdx.x, c = threadIdx.x;
  float v = in[(size_t)n * W + c];
  s1[c] = v; s2[c] = v * v;
  __syncthreads();
  for (int off = 128; off >= 1; off >>= 1) {
    if (c < off) { s1[c] += s1[c + off]; s2[c] += s2[c + off]; }
    __syncthreads();
  }
  float mu  = s1[0] * (1.0f / W);
  float var = s2[0] * (1.0f / W) - mu * mu;
  float nv  = (v - mu) * rsqrtf(var + 1e-5f);
  out[(size_t)n * W + c] = nv;
  if (outh) outh[(size_t)n * W + c] = (f16)nv;
}

// ---- edge prep: edge embed, xj+ee (f16 GEMM operand), per-head attention ---
__global__ void __launch_bounds__(256) k_edge_prep(
    const float* __restrict__ h, const int* __restrict__ src,
    const int* __restrict__ dst, const int* __restrict__ ea,
    const float* __restrict__ e0, const float* __restrict__ e1a,
    const float* __restrict__ e1b, const float* __restrict__ ez,
    f16* __restrict__ xjh, float* __restrict__ att)
{
  __shared__ float s[256];
  int e = blockIdx.x, c = threadIdx.x;
  int a0 = ea[(size_t)e * 3 + 0] + 1;
  int a1 = ea[(size_t)e * 3 + 1] + 1;
  int a2 = ea[(size_t)e * 3 + 2] + 1;
  float eev = e0[(size_t)a0 * W + c] +
              0.5f * (e1a[(size_t)a1 * W + c] * expf(ez[0]) +
                      e1b[(size_t)a2 * W + c] * expf(ez[1]));
  float xj = h[(size_t)src[e] * W + c] + eev;
  float xi = h[(size_t)dst[e] * W + c] + eev;
  xjh[(size_t)e * W + c] = (f16)xj;
  s[c] = xi * xj;
  __syncthreads();
  for (int off = 32; off >= 1; off >>= 1) {   // reduce each 64-wide head
    if ((c & 63) < off) s[c] += s[c + off];
    __syncthreads();
  }
  if ((c & 63) == 0)
    att[(size_t)e * 4 + (c >> 6)] = 2.0f / (1.0f + expf(-s[c]));
}

// ---- segment sum (atomic), width = 256 -------------------------------------
__global__ void __launch_bounds__(256) k_segpool(
    const float* __restrict__ v, const int* __restrict__ seg,
    float* __restrict__ pool)
{
  int n = blockIdx.x, c = threadIdx.x;
  atomicAdd(pool + (size_t)seg[n] * W + c, v[(size_t)n * W + c]);
}

// ---- combine: t += pmsg[batch]*exp(z)  (glob path), writes f16 mirror ------
__global__ void __launch_bounds__(256) k_comb_glob(
    float* __restrict__ t, f16* __restrict__ th, const float* __restrict__ pm,
    const int* __restrict__ batch, const float* __restrict__ zero, size_t total)
{
  size_t i = (size_t)blockIdx.x * blockDim.x + threadIdx.x;
  if (i >= total) return;
  float ez = expf(zero[0]);
  size_t n = i >> 9; int c = (int)(i & 511);
  float v = t[i] + pm[((size_t)batch[n] << 9) + c] * ez;
  t[i] = v; th[i] = (f16)v;
}

// ---- combine: t += aggr*exp(z)  (conv path), writes f16 mirror -------------
__global__ void __launch_bounds__(256) k_comb_conv(
    float* __restrict__ t, f16* __restrict__ th, const float* __restrict__ a,
    const float* __restrict__ zero, size_t total)
{
  size_t i = (size_t)blockIdx.x * blockDim.x + threadIdx.x;
  if (i >= total) return;
  float v = t[i] + a[i] * expf(zero[0]);
  t[i] = v; th[i] = (f16)v;
}

// ---- f32 -> f16 elementwise ------------------------------------------------
__global__ void __launch_bounds__(256) k_cvt(
    const float* __restrict__ in, f16* __restrict__ out, size_t n)
{
  size_t i = (size_t)blockIdx.x * blockDim.x + threadIdx.x;
  if (i < n) out[i] = (f16)in[i];
}

// ---- head0: out[g] = dot(xglob[g,:], w) + b --------------------------------
__global__ void __launch_bounds__(256) k_head0(
    const float* __restrict__ xg, const float* __restrict__ w,
    const float* __restrict__ b, float* __restrict__ out)
{
  __shared__ float s[256];
  int g = blockIdx.x, c = threadIdx.x;
  s[c] = xg[(size_t)g * W + c] * w[c];
  __syncthreads();
  for (int off = 128; off >= 1; off >>= 1) {
    if (c < off) s[c] += s[c + off];
    __syncthreads();
  }
  if (c == 0) out[g] = s[0] + b[0];
}

} // namespace

// ============================================================================
// Host orchestration
// ============================================================================
extern "C" void kernel_launch(void* const* d_in, const int* in_sizes, int n_in,
                              void* d_out, int out_size, void* d_ws, size_t ws_size,
                              hipStream_t stream) {
  (void)in_sizes; (void)n_in; (void)out_size; (void)ws_size;
  auto F = [&](int i) { return (const float*)d_in[i]; };
  const int* x    = (const int*)d_in[0];
  const int* src  = (const int*)d_in[1];            // edge_index[0]
  const int* dst  = ((const int*)d_in[1]) + N_EDGES;// edge_index[1]
  const int* ea   = (const int*)d_in[2];
  const int* bat  = (const int*)d_in[3];
  // params flattened in insertion order: embed(e0, e1[0..7], zero) = 4..13,
  // then per layer (base 14 + 28*layer):
  //  +0 glob.zero +1/2 pre.w/b +3/4 mem +5/6 msg +7/8 post
  //  +9 conv.e0 +10/11 conv.e1 +12 conv.embed.zero +13 conv.zero
  //  +14/15 pre +16/17 mem +18/19 msg +20/21 post
  //  +22..27 dense l1/l2/l3 (w,b)
  // then 70/71 post.w/b, 72/73 head0, 74/75 head1.

  float* out_x0 = (float*)d_out;
  float* out_x1 = out_x0 + N_GRAPHS;
  float* out_x2 = out_x1 + (size_t)N_NODES * OUT_X1;

  // ---- workspace bump allocator -------------------------------------------
  char* wp = (char*)d_ws;
  auto alloc = [&](size_t bytes) -> void* {
    void* r = wp; wp += (bytes + 255) & ~(size_t)255; return r;
  };
  float* xx     = (float*)alloc((size_t)N_NODES * W  * 4);
  f16*   xxh    = (f16*)  alloc((size_t)N_NODES * W  * 2);
  float* h      = (float*)alloc((size_t)N_NODES * W  * 4);
  f16*   hh     = (f16*)  alloc((size_t)N_NODES * W  * 2);
  float* t512   = (float*)alloc((size_t)N_NODES * W2 * 4);
  f16*   t512h  = (f16*)  alloc((size_t)N_NODES * W2 * 2);
  float* aggr   = (float*)alloc((size_t)N_NODES * W2 * 4);
  float* att    = (float*)alloc((size_t)N_EDGES * 4  * 4);
  float* pooled = (float*)alloc((size_t)N_GRAPHS * W  * 4);
  f16*   pooledh= (f16*)  alloc((size_t)N_GRAPHS * W  * 2);
  float* pmsg   = (float*)alloc((size_t)N_GRAPHS * W2 * 4);
  float* xglob  = (float*)alloc((size_t)N_GRAPHS * W  * 4);
  f16*   xjh    = (f16*)t512;   // alias: edge operand live only before t512 reuse
  float* xres   = aggr;         // alias: dense residual after conv aggr is done

  // ---- weight conversion (f32 [K,Nout] -> f16 [NoutPad,K]) ----------------
  auto cvtw = [&](int idx, int K, int Nout, int NoutPad) -> f16* {
    f16* dstp = (f16*)alloc((size_t)NoutPad * K * 2);
    int n = K * NoutPad;
    k_wt<<<(n + 255) / 256, 256, 0, stream>>>(F(idx), dstp, K, Nout, NoutPad);
    return dstp;
  };
  f16 *wg_pre[DEPTH], *wg_mem[DEPTH], *wg_msg[DEPTH], *wg_post[DEPTH];
  f16 *wc_pre[DEPTH], *wc_mem[DEPTH], *wc_msg[DEPTH], *wc_post[DEPTH];
  f16 *wd_l1[DEPTH],  *wd_l2[DEPTH],  *wd_l3[DEPTH];
  for (int i = 0; i < DEPTH; ++i) {
    int b = 14 + 28 * i;
    wg_pre[i]  = cvtw(b + 1,  W,  W,  W);
    wg_mem[i]  = cvtw(b + 3,  W,  W2, W2);
    wg_msg[i]  = cvtw(b + 5,  W,  W2, W2);
    wg_post[i] = cvtw(b + 7,  W2, W,  W);
    wc_pre[i]  = cvtw(b + 14, W,  W,  W);
    wc_mem[i]  = cvtw(b + 16, W,  W2, W2);
    wc_msg[i]  = cvtw(b + 18, W,  W2, W2);
    wc_post[i] = cvtw(b + 20, W2, W,  W);
    wd_l1[i]   = cvtw(b + 22, W,  W,  W);
    wd_l2[i]   = cvtw(b + 24, W,  W2, W2);
    wd_l3[i]   = cvtw(b + 26, W2, W,  W);
  }
  f16* w_post  = cvtw(70, W, W, W);
  f16* w_head1 = cvtw(74, W, OUT_X1, 192);   // pad Nout 174 -> 192 (zeros)

  // ---- GEMM launcher (32x64 tile per wave) --------------------------------
  auto gemm = [&](int mode, const f16* A, const f16* Wt, const float* bias,
                  float* out, f16* outh, float* out2, const float* attp,
                  const int* dsti, float* scat,
                  int M, int Ncols, int Ntile, int K, int ldc) {
    int waves  = (M >> 5) * (Ntile >> 6);
    int blocks = (waves + 7) / 8;
    switch (mode) {
      case MODE_PLAIN: k_gemm<MODE_PLAIN><<<blocks, 256, 0, stream>>>(
          A, Wt, bias, out, outh, out2, attp, dsti, scat, M, Ncols, Ntile, K, ldc); break;
      case MODE_GELU:  k_gemm<MODE_GELU><<<blocks, 256, 0, stream>>>(
          A, Wt, bias, out, outh, out2, attp, dsti, scat, M, Ncols, Ntile, K, ldc); break;
      case MODE_ADD:   k_gemm<MODE_ADD><<<blocks, 256, 0, stream>>>(
          A, Wt, bias, out, outh, out2, attp, dsti, scat, M, Ncols, Ntile, K, ldc); break;
      default:         k_gemm<MODE_EDGE><<<blocks, 256, 0, stream>>>(
          A, Wt, bias, out, outh, out2, attp, dsti, scat, M, Ncols, Ntile, K, ldc); break;
    }
  };
  size_t tot512 = (size_t)N_NODES * W2;
  int blk512 = (int)(tot512 / 256);

  // ---- embedding -----------------------------------------------------------
  k_embed_nodes<<<N_NODES, 256, 0, stream>>>(
      x, F(4), F(5), F(6), F(7), F(8), F(9), F(10), F(11), F(12), F(13), xx, xxh);

  // ---- layers --------------------------------------------------------------
  for (int i = 0; i < DEPTH; ++i) {
    int b = 14 + 28 * i;
    // ======== GlobBlock ========
    hipMemsetAsync(pooled, 0, (size_t)N_GRAPHS * W * 4, stream);
    gemm(MODE_PLAIN, xxh, wg_pre[i], F(b + 2), h, nullptr, nullptr, nullptr,
         nullptr, nullptr, N_NODES, W, W, W, W);
    k_ln<<<N_NODES, 256, 0, stream>>>(h, h, hh);
    k_segpool<<<N_NODES, 256, 0, stream>>>(h, bat, pooled);
    k_cvt<<<(int)(((size_t)N_GRAPHS * W + 255) / 256), 256, 0, stream>>>(
        pooled, pooledh, (size_t)N_GRAPHS * W);
    gemm(MODE_GELU, pooledh, wg_msg[i], F(b + 6), pmsg, nullptr, nullptr,
         nullptr, nullptr, nullptr, N_GRAPHS, W2, W2, W, W2);
    gemm(MODE_GELU, hh, wg_mem[i], F(b + 4), t512, nullptr, nullptr, nullptr,
         nullptr, nullptr, N_NODES, W2, W2, W, W2);
    k_comb_glob<<<blk512, 256, 0, stream>>>(t512, t512h, pmsg, bat, F(b + 0), tot512);
    gemm(MODE_ADD, t512h, wg_post[i], F(b + 8), xx, xxh, nullptr, nullptr,
         nullptr, nullptr, N_NODES, W, W, W2, W);
    // ======== ConvBlock ========
    hipMemsetAsync(aggr, 0, (size_t)N_NODES * W2 * 4, stream);
    gemm(MODE_PLAIN, xxh, wc_pre[i], F(b + 15), h, nullptr, nullptr, nullptr,
         nullptr, nullptr, N_NODES, W, W, W, W);
    k_ln<<<N_NODES, 256, 0, stream>>>(h, h, hh);
    k_edge_prep<<<N_EDGES, 256, 0, stream>>>(
        h, src, dst, ea, F(b + 9), F(b + 10), F(b + 11), F(b + 12), xjh, att);
    gemm(MODE_EDGE, xjh, wc_msg[i], F(b + 19), nullptr, nullptr, nullptr, att,
         dst, aggr, N_EDGES, W2, W2, W, W2);
    gemm(MODE_GELU, hh, wc_mem[i], F(b + 17), t512, nullptr, nullptr, nullptr,
         nullptr, nullptr, N_NODES, W2, W2, W, W2);   // overwrites xjh alias (done)
    k_comb_conv<<<blk512, 256, 0, stream>>>(t512, t512h, aggr, F(b + 13), tot512);
    gemm(MODE_ADD, t512h, wc_post[i], F(b + 21), xx, xxh, nullptr, nullptr,
         nullptr, nullptr, N_NODES, W, W, W2, W);
    // ======== DenseBlock ========
    gemm(MODE_PLAIN, xxh, wd_l1[i], F(b + 23), h, nullptr, nullptr, nullptr,
         nullptr, nullptr, N_NODES, W, W, W, W);
    k_ln<<<N_NODES, 256, 0, stream>>>(h, h, hh);
    gemm(MODE_GELU, hh, wd_l2[i], F(b + 25), t512, t512h, nullptr, nullptr,
         nullptr, nullptr, N_NODES, W2, W2, W, W2);
    gemm(MODE_ADD, t512h, wd_l3[i], F(b + 27), xx, xxh,
         (i >= DEPTH / 2) ? xres : nullptr, nullptr, nullptr, nullptr,
         N_NODES, W, W, W2, W);
    if (i >= DEPTH / 2)   // xlst has exactly one entry (DEPTH=2) -> mean == LN(xres)
      k_ln<<<N_NODES, 256, 0, stream>>>(xres, out_x2, nullptr);
  }

  // ---- final heads ---------------------------------------------------------
  gemm(MODE_PLAIN, xxh, w_post, F(71), h, nullptr, nullptr, nullptr, nullptr,
       nullptr, N_NODES, W, W, W, W);
  k_ln<<<N_NODES, 256, 0, stream>>>(h, h, hh);          // h = xnode
  gemm(MODE_PLAIN, hh, w_head1, F(75), out_x1, nullptr, nullptr, nullptr,
       nullptr, nullptr, N_NODES, OUT_X1, 192, W, OUT_X1);
  hipMemsetAsync(xglob, 0, (size_t)N_GRAPHS * W * 4, stream);
  k_segpool<<<N_NODES, 256, 0, stream>>>(h, bat, xglob);
  k_head0<<<N_GRAPHS, 256, 0, stream>>>(xglob, F(72), F(73), out_x0);
}
